// SiliconSynapse_3169685865300
// MI455X (gfx1250) — matrix operand order repaired and checked by reading, not compiled, not run on hardware
//
#include <hip/hip_runtime.h>
#include <hip/hip_bf16.h>
#include <math.h>

// ---------------- constants ----------------
#define BB 4096      // batch
#define DD 512       // features
#define EE 32        // experts (== wave32 lanes!)
#define TOPK 4
#define ROWPAD 32                        // expert segment padding (2x2 reg blocking)
#define CAP (BB*TOPK + EE*ROWPAD)        // padded pair-list capacity (17408)
#define ROW_GROUPS_MAX (CAP/ROWPAD)      // 544
#define A_STRIDE 520                     // LDS row stride in bf16 (bank-conflict pad)

typedef __attribute__((ext_vector_type(16))) __bf16 v16bf;
typedef __attribute__((ext_vector_type(8)))  __bf16 v8bf;
typedef __attribute__((ext_vector_type(8)))  float  v8f;

__device__ __forceinline__ float wave_sum(float v) {
#pragma unroll
  for (int o = 16; o; o >>= 1) v += __shfl_xor(v, o, 32);
  return v;
}
__device__ __forceinline__ float wave_max(float v) {
#pragma unroll
  for (int o = 16; o; o >>= 1) v = fmaxf(v, __shfl_xor(v, o, 32));
  return v;
}

// ---------------- K0: init ----------------
__global__ __launch_bounds__(256) void k0_init(float* __restrict__ y,
                                               int* __restrict__ counts,
                                               int* __restrict__ cursors,
                                               int* __restrict__ pairs) {
  int i = blockIdx.x * 256 + threadIdx.x;
  if (i < BB * DD) y[i] = 0.f;
  if (i < EE) { counts[i] = 0; cursors[i] = 0; }
  if (i < CAP) pairs[i] = -1;
}

// ---------------- K1: ternary quantize expert_w -> bf16 + row scale ----------------
__global__ __launch_bounds__(256) void k1_quant(const float* __restrict__ ew,
                                                __bf16* __restrict__ wq,
                                                float* __restrict__ scale) {
  int lane = threadIdx.x & 31, wave = threadIdx.x >> 5;
  int row = blockIdx.x * 8 + wave;            // 0 .. E*D-1 = 16383
  const float* wr = ew + (size_t)row * DD;
  float s = 0.f;
  for (int k = lane; k < DD; k += 32) s += fabsf(wr[k]);
  s = wave_sum(s);
  float sc = fmaxf(s * (1.0f / DD), 1e-5f);
  float inv = 1.0f / sc;
  for (int k = lane; k < DD; k += 32) {
    float q = rintf(wr[k] * inv);             // RNE matches jnp.round
    q = fminf(fmaxf(q, -1.f), 1.f);           // ternary: exact in bf16
    wq[(size_t)row * DD + k] = (__bf16)q;
  }
  if (lane == 0) scale[row] = sc;
}

// ---------------- K1b: alpha = clip(DT / (softplus(tau)+1e-3), 0, 1) ----------------
__global__ __launch_bounds__(256) void k1b_alpha(const float* __restrict__ raw_tau,
                                                 float* __restrict__ alpha) {
  int i = blockIdx.x * 256 + threadIdx.x;
  if (i >= EE * DD) return;
  float t = raw_tau[i];
  float sp = (t > 20.f) ? t : log1pf(__expf(t));
  float a = 0.02f / (sp + 1e-3f);
  alpha[i] = fminf(fmaxf(a, 0.f), 1.f);
}

// ---------------- K1c: pre-convert x to bf16 (once, instead of per-tile) ----------------
__global__ __launch_bounds__(256) void k1c_xcast(const float* __restrict__ x,
                                                 __bf16* __restrict__ xbf) {
  int i = blockIdx.x * 256 + threadIdx.x;
  if (i < BB * DD) xbf[i] = (__bf16)x[i];
}

// ---------------- K2: router, one wave per token, lane == expert ----------------
__global__ __launch_bounds__(256) void k2_router(const float* __restrict__ x,
                                                 const float* __restrict__ pe,
                                                 const float* __restrict__ up,
                                                 const float* __restrict__ gw,
                                                 const float* __restrict__ gb,
                                                 const unsigned char* __restrict__ alive,
                                                 float* __restrict__ topw,
                                                 int* __restrict__ topidx,
                                                 int* __restrict__ counts) {
  int lane = threadIdx.x & 31, wave = threadIdx.x >> 5;
  int b = blockIdx.x * 8 + wave;
  if (b >= BB) return;
  const float* xb = x + (size_t)b * DD;
  const float* gr = gw + (size_t)lane * DD;     // lane = expert
  float acc = 0.f;
  for (int k = 0; k < DD; ++k) acc = fmaf(xb[k], gr[k], acc);
  float u = up[lane];
  float logit = acc + gb[lane] + pe[b] * (1.f - u) - u;  // novelty & usage scales = 1
  if (!alive[lane]) logit = -3.402823466e38f;
  // wave softmax over 32 experts
  float m = wave_max(logit);
  float ex = __expf(logit - m);
  float denom = wave_sum(ex);
  float prob = ex / denom;
  // top-4 via repeated wave argmax (lowest-index tie-break, like lax.top_k)
  float pv = prob;
  float wsel[TOPK]; int isel[TOPK];
#pragma unroll
  for (int r = 0; r < TOPK; ++r) {
    float bv = pv; int bi = lane;
#pragma unroll
    for (int o = 16; o; o >>= 1) {
      float ov = __shfl_xor(bv, o, 32);
      int   oi = __shfl_xor(bi, o, 32);
      if (ov > bv || (ov == bv && oi < bi)) { bv = ov; bi = oi; }
    }
    wsel[r] = bv; isel[r] = bi;
    if (lane == bi) pv = -3.402823466e38f;
  }
  float tot = fmaxf(wsel[0] + wsel[1] + wsel[2] + wsel[3], 1e-6f);
  if (lane < TOPK) {
    topw[b * TOPK + lane]   = wsel[lane] / tot;
    topidx[b * TOPK + lane] = isel[lane];
    atomicAdd(&counts[isel[lane]], 1);
  }
}

// ---------------- K3: per-expert offsets, padded to ROWPAD ----------------
__global__ void k3_scan(const int* __restrict__ counts, int* __restrict__ offsets) {
  if (threadIdx.x == 0 && blockIdx.x == 0) {
    int off = 0;
    for (int e = 0; e < EE; ++e) {
      offsets[e] = off;
      off += ((counts[e] + ROWPAD - 1) / ROWPAD) * ROWPAD;
    }
    offsets[EE] = off;
  }
}

// ---------------- K4: scatter (b,k) pairs into expert segments ----------------
__global__ __launch_bounds__(256) void k4_scatter(const int* __restrict__ topidx,
                                                  const int* __restrict__ offsets,
                                                  int* __restrict__ cursors,
                                                  int* __restrict__ pairs) {
  int p = blockIdx.x * 256 + threadIdx.x;   // p = b*4 + k
  if (p >= BB * TOPK) return;
  int e = topidx[p];
  int pos = atomicAdd(&cursors[e], 1);
  pairs[offsets[e] + pos] = p;
}

// ---------------- K5: grouped bf16 WMMA GEMM (2x2 reg-blocked) + liquid update ----------------
__global__ __launch_bounds__(256) void k5_moe(const __bf16* __restrict__ xbf,
                                              const float* __restrict__ hidden,
                                              const __bf16* __restrict__ wq,
                                              const float* __restrict__ scale,
                                              const float* __restrict__ alpha,
                                              const float* __restrict__ topw,
                                              const int* __restrict__ pairs,
                                              const int* __restrict__ offsets,
                                              float* __restrict__ y) {
  __shared__ __bf16 ldsA[32 * A_STRIDE];
  __shared__ int spairs[32];
  __shared__ int sexpert;

  int tid = threadIdx.x, lane = tid & 31, wave = tid >> 5;
  int base_row = blockIdx.y * ROWPAD;
  int total = offsets[EE];
  if (base_row >= total) return;            // uniform across block

  if (tid == 0) {
    int e = 0;
    while (e < EE - 1 && offsets[e + 1] <= base_row) ++e;
    sexpert = e;
  }
  if (tid < 32) spairs[tid] = pairs[base_row + tid];
  __syncthreads();
  int e = sexpert;

  // stage 32-token bf16 x tile into LDS: pure 16-byte copies (32 rows x 64 chunks)
  for (int idx = tid; idx < 32 * 64; idx += 256) {
    int row = idx >> 6, chunk = idx & 63;
    int pr = spairs[row];
    v8bf v = {(__bf16)0.f, (__bf16)0.f, (__bf16)0.f, (__bf16)0.f,
              (__bf16)0.f, (__bf16)0.f, (__bf16)0.f, (__bf16)0.f};
    if (pr >= 0) v = *(const v8bf*)(xbf + (size_t)(pr >> 2) * DD + chunk * 8);
    *(v8bf*)(ldsA + row * A_STRIDE + chunk * 8) = v;
  }
  __syncthreads();

  int half = lane >> 4, nl = lane & 15;
  int n0 = blockIdx.x * 256 + wave * 32 + nl;           // global out-feature (tile 0)
  // B fragment: lane holds wq[e][n][k..k+15] contiguous (halves split K 0-15 / 16-31)
  const __bf16* bptr0 = wq + ((size_t)e * DD + n0) * DD + half * 16;
  const __bf16* bptr1 = bptr0 + (size_t)16 * DD;        // n0+16
  // A fragment: lane&15 = M row; half selects K 0-15 vs 16-31
  const __bf16* aptr0 = ldsA + nl * A_STRIDE + half * 16;
  const __bf16* aptr1 = aptr0 + 16 * A_STRIDE;          // rows 16..31

  v8f acc[2][2];
#pragma unroll
  for (int i = 0; i < 2; ++i)
#pragma unroll
    for (int j = 0; j < 2; ++j)
      acc[i][j] = (v8f){0.f, 0.f, 0.f, 0.f, 0.f, 0.f, 0.f, 0.f};

#pragma unroll 2
  for (int kk = 0; kk < DD; kk += 32) {
    union { v16bf v; v8bf h[2]; } a0, a1;
    a0.h[0] = *(const v8bf*)(aptr0 + kk);        // K = kk + 16*half .. +7
    a0.h[1] = *(const v8bf*)(aptr0 + kk + 8);    // K = kk + 16*half + 8 .. +15
    a1.h[0] = *(const v8bf*)(aptr1 + kk);
    a1.h[1] = *(const v8bf*)(aptr1 + kk + 8);
    v16bf b0 = *(const v16bf*)(bptr0 + kk);      // K = kk + 16*half .. +15
    v16bf b1 = *(const v16bf*)(bptr1 + kk);
    acc[0][0] = __builtin_amdgcn_wmma_f32_16x16x32_bf16(false, a0.v, false, b0,
                                                        (short)0, acc[0][0], false, false);
    acc[0][1] = __builtin_amdgcn_wmma_f32_16x16x32_bf16(false, a0.v, false, b1,
                                                        (short)0, acc[0][1], false, false);
    acc[1][0] = __builtin_amdgcn_wmma_f32_16x16x32_bf16(false, a1.v, false, b0,
                                                        (short)0, acc[1][0], false, false);
    acc[1][1] = __builtin_amdgcn_wmma_f32_16x16x32_bf16(false, a1.v, false, b1,
                                                        (short)0, acc[1][1], false, false);
  }

#pragma unroll
  for (int nt = 0; nt < 2; ++nt) {
    int n = n0 + nt * 16;
    float sc = scale[e * DD + n];
    float h  = hidden[e * DD + n];
    float al = alpha[e * DD + n];
#pragma unroll
    for (int mt = 0; mt < 2; ++mt) {
      v8f c = acc[mt][nt];
#pragma unroll
      for (int r = 0; r < 8; ++r) {
        int m = mt * 16 + r + 8 * half;          // C layout: VGPR r -> M = r + 8*(lane>>4)
        int pr = spairs[m];
        if (pr < 0) continue;
        float w = topw[pr];                      // pr = b*4 + k indexes top_w directly
        float pre = c[r] * sc + h;               // scale factors out of ternary K-sum
        float s = tanhf(pre);
        float outv = h + al * (s - h);
        unsafeAtomicAdd(&y[(size_t)(pr >> 2) * DD + n], w * outv);
      }
    }
  }
}

// ---------------- launch ----------------
extern "C" void kernel_launch(void* const* d_in, const int* in_sizes, int n_in,
                              void* d_out, int out_size, void* d_ws, size_t ws_size,
                              hipStream_t stream) {
  const float* x     = (const float*)d_in[0];
  const float* hidden= (const float*)d_in[1];
  const float* pe    = (const float*)d_in[2];
  const float* up    = (const float*)d_in[3];
  const float* gw    = (const float*)d_in[4];
  const float* gb    = (const float*)d_in[5];
  const float* ew    = (const float*)d_in[6];
  const float* rtau  = (const float*)d_in[7];
  const unsigned char* alive = (const unsigned char*)d_in[8];
  float* y = (float*)d_out;

  char* w = (char*)d_ws;
  __bf16* wq   = (__bf16*)w;  w += (size_t)EE * DD * DD * sizeof(__bf16);
  __bf16* xbf  = (__bf16*)w;  w += (size_t)BB * DD * sizeof(__bf16);
  float* scale = (float*)w;   w += (size_t)EE * DD * sizeof(float);
  float* alpha = (float*)w;   w += (size_t)EE * DD * sizeof(float);
  float* topw  = (float*)w;   w += (size_t)BB * TOPK * sizeof(float);
  int* topidx  = (int*)w;     w += (size_t)BB * TOPK * sizeof(int);
  int* counts  = (int*)w;     w += 128;
  int* cursors = (int*)w;     w += 128;
  int* offsets = (int*)w;     w += 256;
  int* pairs   = (int*)w;     w += (size_t)CAP * sizeof(int);

  k0_init<<<(BB * DD + 255) / 256, 256, 0, stream>>>(y, counts, cursors, pairs);
  k1_quant<<<(EE * DD) / 8, 256, 0, stream>>>(ew, wq, scale);
  k1b_alpha<<<(EE * DD + 255) / 256, 256, 0, stream>>>(rtau, alpha);
  k1c_xcast<<<(BB * DD + 255) / 256, 256, 0, stream>>>(x, xbf);
  k2_router<<<BB / 8, 256, 0, stream>>>(x, pe, up, gw, gb, alive, topw, topidx, counts);
  k3_scan<<<1, 32, 0, stream>>>(counts, offsets);
  k4_scatter<<<(BB * TOPK) / 256, 256, 0, stream>>>(topidx, offsets, cursors, pairs);
  dim3 g5(DD / 256, ROW_GROUPS_MAX, 1);
  k5_moe<<<g5, 256, 0, stream>>>(xbf, hidden, wq, scale, alpha, topw, pairs, offsets, y);
}
